// TreeWmodel_40020505264428
// MI455X (gfx1250) — matrix operand order repaired and checked by reading, not compiled, not run on hardware
//
#include <hip/hip_runtime.h>
#include <hip/hip_bf16.h>
#include <math.h>

// CDNA5 / gfx1250, wave32. One wave handles 16 (b,n) pairs.
// xs - ys = A * (softmax(emb[x]) - softmax(emb[y])), A = tree ancestor matrix.
// A factors into 17 16x16 tiles (16 depth-4..7 subtrees + top tree + totals),
// each evaluated with v_wmma_f32_16x16x32_f16 (f16 A is exact 0/1, f32 accum).

typedef _Float16 v16h __attribute__((ext_vector_type(16)));
typedef float    v8f  __attribute__((ext_vector_type(8)));

#define E_DIM 255

// 1-based heap ancestor-or-self test: is p1 an ancestor of (or equal to) c1?
__device__ __forceinline__ bool anc1(unsigned p1, unsigned c1) {
  int dc = (31 - __clz((int)c1)) - (31 - __clz((int)p1));
  return (dc >= 0) && ((c1 >> dc) == p1);
}

// Softmax of a 255-float row; lane k owns elements {lane, lane+32, ...}.
__device__ __forceinline__ void softmax_row(const float* __restrict__ row,
                                            int lane, float p[8]) {
  float v[8];
  float m = -INFINITY;
#pragma unroll
  for (int k = 0; k < 8; ++k) {
    int idx = k * 32 + lane;
    v[k] = (idx < E_DIM) ? row[idx] : -INFINITY;
    m = fmaxf(m, v[k]);
  }
#pragma unroll
  for (int off = 16; off >= 1; off >>= 1)
    m = fmaxf(m, __shfl_xor(m, off, 32));
  float s = 0.f;
#pragma unroll
  for (int k = 0; k < 8; ++k) {
    int idx = k * 32 + lane;
    float e = (idx < E_DIM) ? __expf(v[k] - m) : 0.f;
    p[k] = e;
    s += e;
  }
#pragma unroll
  for (int off = 16; off >= 1; off >>= 1)
    s += __shfl_xor(s, off, 32);
  float inv = 1.0f / s;
#pragma unroll
  for (int k = 0; k < 8; ++k) p[k] *= inv;
}

__global__ void __launch_bounds__(32)
treewe_kernel(const int* __restrict__ x, const int* __restrict__ y,
              const float* __restrict__ emb, const float* __restrict__ w,
              float* __restrict__ out, int npairs) {
  // B-operand staging: [block][pair-column][K=0..31] f16, K>=16 stays zero.
  __shared__ __align__(32) _Float16 tile[17][16][32];
  __shared__ float wtile[17][16];  // |weight| remapped to (block, local-row)

  const int lane = threadIdx.x;       // 0..31 (wave32)
  const int g    = blockIdx.x;        // 16-pair group id

  // ---- zero the tile (incl. the K=16..31 padding half) ----
  unsigned int* tz = (unsigned int*)&tile[0][0][0];
  for (int i = lane; i < (17 * 16 * 32) / 2; i += 32) tz[i] = 0u;

  // ---- weight tile: wtile[b][M] = |w[global(b,M)]|, row 15 = 0 ----
  for (int i = lane; i < 17 * 16; i += 32) {
    int b = i >> 4, M = i & 15;
    float wv = 0.f;
    if (M < 15) {
      int gidx;
      if (b == 16) {
        gidx = M;  // top tree: nodes 0..14
      } else {
        unsigned l1 = (unsigned)(M + 1);            // 1-based local
        int dl = 31 - __clz((int)l1);               // local depth 0..3
        gidx = (int)(((unsigned)(16 + b) << dl) + (l1 - (1u << dl))) - 1;
      }
      wv = fabsf(w[gidx]);
    }
    wtile[b][M] = wv;
  }

  // ---- build A15 (15-node subtree ancestor matrix) and Anc (top x totals)
  //      in f16 WMMA A-operand layout: lane M=0..15, v16h element e -> K ----
  const int Mrow = lane & 15;
  const int hiK  = (lane >= 16) ? 8 : 0;
  v16h a15, a2;
#pragma unroll
  for (int e = 0; e < 16; ++e) {
    int vv = e >> 1, h = e & 1;
    int K = (hiK << 1) + e;  // 0..31
    float f15 = (Mrow < 15 && K < 15 && anc1((unsigned)(Mrow + 1), (unsigned)(K + 1))) ? 1.f : 0.f;
    float f2  = (Mrow < 15 && K < 16 && anc1((unsigned)(Mrow + 1), (unsigned)(16 + K))) ? 1.f : 0.f;
    a15[e] = (_Float16)f15;
    a2[e]  = (_Float16)f2;
  }

  // ---- per pair: gather rows, softmax, scatter f16 diff into tile ----
  for (int q = 0; q < 16; ++q) {
    int pair = g * 16 + q;
    if (pair >= npairs) break;  // uniform across the wave
    const float* rx = emb + (size_t)x[pair] * E_DIM;
    const float* ry = emb + (size_t)y[pair] * E_DIM;
    __builtin_prefetch(ry, 0, 0);  // global_prefetch_b8 of the second row
    float px[8], py[8];
    softmax_row(rx, lane, px);
    softmax_row(ry, lane, py);
#pragma unroll
    for (int k = 0; k < 8; ++k) {
      int idx = k * 32 + lane;
      if (idx < E_DIM) {
        float sv = px[k] - py[k];
        int blk, loc;
        if (idx < 15) {                 // top tree node
          blk = 16; loc = idx;
        } else {                        // map to depth-4 subtree + local heap idx
          unsigned n = (unsigned)(idx + 1);
          int d  = 31 - __clz((int)n);  // depth 4..7
          int sh = d - 4;
          unsigned r1 = n >> sh;        // 1-based depth-4 root: 16..31
          blk = (int)r1 - 16;
          loc = (int)(n - (r1 << sh)) + (1 << sh) - 1;  // 0..14
        }
        tile[blk][q][loc] = (_Float16)sv;
      }
    }
  }
  __syncthreads();  // single-wave WG: cheap; orders LDS scatter vs. B loads

  // ---- WMMA phase ----
  const int bhalf = (lane >= 16) ? 16 : 0;  // B layout: lanes 16..31 hold K=16..31
  const int col   = lane & 15;              // B/C column = pair index
  float tval[16];                           // subtree totals, B-layout in-register
  float sumabs = 0.f;

#pragma unroll
  for (int b = 0; b < 16; ++b) {
    v16h bv = *(const v16h*)&tile[b][col][bhalf];
    v8f c = {};
    c = __builtin_amdgcn_wmma_f32_16x16x32_f16(false, a15, false, bv,
                                               (short)0, c, false, false);
    // row 0 of block result = subtree total; lanes>=16 pad K>=16 with zeros
    tval[b] = (lane < 16) ? c[0] : 0.f;
#pragma unroll
    for (int r = 0; r < 8; ++r)
      sumabs += fabsf(c[r]) * wtile[b][r + hiK];
  }

  {  // top tree: A15 * s_top + Anc * totals
    v16h bv = *(const v16h*)&tile[16][col][bhalf];
    v8f c = {};
    c = __builtin_amdgcn_wmma_f32_16x16x32_f16(false, a15, false, bv,
                                               (short)0, c, false, false);
    v16h tv;
#pragma unroll
    for (int e = 0; e < 16; ++e) tv[e] = (_Float16)tval[e];
    c = __builtin_amdgcn_wmma_f32_16x16x32_f16(false, a2, false, tv,
                                               (short)0, c, false, false);
#pragma unroll
    for (int r = 0; r < 8; ++r)
      sumabs += fabsf(c[r]) * wtile[16][r + hiK];
  }

  // combine the two row-halves (lane N holds rows 0-7, lane N+16 rows 8-15)
  float tot = sumabs + __shfl_xor(sumabs, 16, 32);
  int pair = g * 16 + col;
  if (lane < 16 && pair < npairs) out[pair] = -tot;
}

extern "C" void kernel_launch(void* const* d_in, const int* in_sizes, int n_in,
                              void* d_out, int out_size, void* d_ws, size_t ws_size,
                              hipStream_t stream) {
  const int*   x   = (const int*)d_in[0];
  const int*   y   = (const int*)d_in[1];
  const float* emb = (const float*)d_in[2];
  const float* w   = (const float*)d_in[3];
  float* out = (float*)d_out;
  int P = in_sizes[0];                 // 4096*50 pairs
  int groups = (P + 15) / 16;          // one wave32 per 16 pairs
  treewe_kernel<<<groups, 32, 0, stream>>>(x, y, emb, w, out, P);
}
